// GTRBlock_87067577025249
// MI455X (gfx1250) — compile-verified, matrix operand order checked by reading, not run on hardware
//
#include <hip/hip_runtime.h>
#include <hip/hip_bf16.h>

typedef __attribute__((ext_vector_type(16))) _Float16 v16h;
typedef __attribute__((ext_vector_type(8)))  _Float16 v8h;
typedef __attribute__((ext_vector_type(4)))  _Float16 v4h;
typedef __attribute__((ext_vector_type(8)))  float    v8f;

union AFrag { unsigned u[8]; v16h h; };

constexpr int   Bc = 4, Tc = 4096, Dc = 1024, NTc = 8, Hc = 8192;
constexpr int   BT = Bc * Tc;               // 16384 sequence rows
constexpr float EPSc   = 1e-6f;
constexpr float SCALEc = 0.03125f;          // 1/sqrt(1024)
constexpr int   TCHUNK = 256;               // ctx1 partial chunk
constexpr int   NCH    = Tc / TCHUNK;       // 16 chunks

// ---------------------------------------------------------------------------
// Pack fp32 KxN weight into WMMA B-fragment order as f16.
// Frag (n_tile, k_tile) = 512 halves; per-lane 16 contiguous halves (32B).
// Within a lane: VGPR r holds rows k0=16*(r>>2)+8*(lane>>4)+2*(r&3), k0+1.
// ---------------------------------------------------------------------------
__global__ void pack_weight(const float* __restrict__ W, _Float16* __restrict__ Wp,
                            int K, int N) {
  int idx = blockIdx.x * 256 + threadIdx.x;
  if (idx >= K * N) return;
  int k = idx / N, n = idx - k * N;
  int kk = k & 31, ktile = k >> 5, ntile = n >> 4;
  int blk = kk >> 4;          // which 16-row block
  int t   = kk & 15;
  int grp = t >> 3;           // lane half (0: lanes 0-15, 1: lanes 16-31)
  int t2  = t & 7;
  int r   = blk * 4 + (t2 >> 1);
  int h   = t2 & 1;
  int lane = (n & 15) + (grp << 4);
  size_t off = (((size_t)ntile * (K >> 5) + ktile) << 9) + ((size_t)lane << 4) + (r << 1) + h;
  Wp[off] = (_Float16)W[idx];
}

// ---------------------------------------------------------------------------
// RMSNorm one row of x (D=1024) with two weight vectors -> two f16 outputs.
// ---------------------------------------------------------------------------
__global__ void rms_cast2(const float* __restrict__ x, const float* __restrict__ w1,
                          const float* __restrict__ w2, _Float16* __restrict__ s1,
                          _Float16* __restrict__ s2) {
  const int row = blockIdx.x, tid = threadIdx.x;
  const float4 xv = ((const float4*)(x + (size_t)row * Dc))[tid];
  __shared__ float red[256];
  red[tid] = xv.x * xv.x + xv.y * xv.y + xv.z * xv.z + xv.w * xv.w;
  __syncthreads();
  for (int s = 128; s > 0; s >>= 1) { if (tid < s) red[tid] += red[tid + s]; __syncthreads(); }
  const float inv = rsqrtf(red[0] * (1.0f / Dc) + EPSc);
  const float4 w1v = ((const float4*)w1)[tid];
  const float4 w2v = ((const float4*)w2)[tid];
  v4h o1, o2;
  o1[0] = (_Float16)(xv.x * w1v.x * inv); o1[1] = (_Float16)(xv.y * w1v.y * inv);
  o1[2] = (_Float16)(xv.z * w1v.z * inv); o1[3] = (_Float16)(xv.w * w1v.w * inv);
  o2[0] = (_Float16)(xv.x * w2v.x * inv); o2[1] = (_Float16)(xv.y * w2v.y * inv);
  o2[2] = (_Float16)(xv.z * w2v.z * inv); o2[3] = (_Float16)(xv.w * w2v.w * inv);
  ((v4h*)(s1 + (size_t)row * Dc))[tid] = o1;
  ((v4h*)(s2 + (size_t)row * Dc))[tid] = o2;
}

// ---------------------------------------------------------------------------
// RMSNorm fp32 -> fp32 (small global-token rows).
// ---------------------------------------------------------------------------
__global__ void rms_rows(const float* __restrict__ in, const float* __restrict__ w,
                         float* __restrict__ out) {
  const int row = blockIdx.x, tid = threadIdx.x;
  const float4 xv = ((const float4*)(in + (size_t)row * Dc))[tid];
  __shared__ float red[256];
  red[tid] = xv.x * xv.x + xv.y * xv.y + xv.z * xv.z + xv.w * xv.w;
  __syncthreads();
  for (int s = 128; s > 0; s >>= 1) { if (tid < s) red[tid] += red[tid + s]; __syncthreads(); }
  const float inv = rsqrtf(red[0] * (1.0f / Dc) + EPSc);
  const float4 wv = ((const float4*)w)[tid];
  float4 o;
  o.x = xv.x * wv.x * inv; o.y = xv.y * wv.y * inv;
  o.z = xv.z * wv.z * inv; o.w = xv.w * wv.w * inv;
  ((float4*)(out + (size_t)row * Dc))[tid] = o;
}

// ---------------------------------------------------------------------------
// WMMA GEMM: C[M,N] = A[M,K] (f16 row-major) * Wp (pre-packed f16 frags)
//            + bias[N]  (+ resid fp32 for the fp32-output path).
// Block 256 thr = 8 waves (2x4), block tile 128x256, wave tile 64x64 (4x4
// 16x16 C tiles -> each A/B frag feeds 4 WMMAs), K-step 64.
// ---------------------------------------------------------------------------
__global__ __launch_bounds__(256) void gemm_wmma(
    const _Float16* __restrict__ A, const _Float16* __restrict__ Bp,
    const float* __restrict__ bias, const float* __restrict__ resid,
    _Float16* __restrict__ outH, float* __restrict__ outF,
    int M, int N, int K) {
  constexpr int BM = 128, BN = 256, BK = 64, ASTRIDE = 72;  // 144B row, 16B aligned
  __shared__ _Float16 As[BM * ASTRIDE];
  const int tid  = threadIdx.x;
  const int lane = tid & 31, wave = tid >> 5;
  const int wRow = wave >> 2, wCol = wave & 3;    // 2 x 4 wave grid
  const int mBase = blockIdx.x * BM;
  const int nBase = blockIdx.y * BN;
  const int kTiles = K >> 5;

  const v8f zero8 = {0.f, 0.f, 0.f, 0.f, 0.f, 0.f, 0.f, 0.f};
  v8f acc[4][4];
#pragma unroll
  for (int mt = 0; mt < 4; ++mt)
#pragma unroll
    for (int nt = 0; nt < 4; ++nt) acc[mt][nt] = zero8;

  // A staging: 128 rows x 64 halves (16KB), 64B per thread.
  const int lrow = tid >> 1;            // 0..127
  const int lcol = (tid & 1) << 5;      // 0 or 32 halves
  const _Float16* Aload  = A + (size_t)(mBase + lrow) * K + lcol;
  _Float16*       Astore = &As[lrow * ASTRIDE + lcol];

  const int mLane  = lane & 15;
  const int khalf8 = (lane >> 4) << 3;  // 8-col select for A layout

  for (int k0 = 0; k0 < K; k0 += BK) {
    const v8h* src = (const v8h*)(Aload + k0);
    ((v8h*)Astore)[0] = src[0];
    ((v8h*)Astore)[1] = src[1];
    ((v8h*)Astore)[2] = src[2];
    ((v8h*)Astore)[3] = src[3];
    if (k0 + BK < K) __builtin_prefetch(Aload + k0 + BK, 0, 1);  // global_prefetch_b8
    __syncthreads();
#pragma unroll
    for (int kt = 0; kt < 2; ++kt) {
      const int kk = kt << 5;
      AFrag a[4];
#pragma unroll
      for (int mt = 0; mt < 4; ++mt) {
        const _Float16* ar = &As[(wRow * 64 + mt * 16 + mLane) * ASTRIDE + kk + khalf8];
#pragma unroll
        for (int r = 0; r < 8; ++r) {
          const int kOff = ((r & 4) << 2) + ((r & 3) << 1);   // 16*(r>>2) + 2*(r&3)
          a[mt].u[r] = *(const unsigned*)&ar[kOff];           // ds_load_b128 after vectorize
        }
      }
      const size_t ktile  = (size_t)((k0 >> 5) + kt);
      const int    ntBase = (nBase >> 4) + (wCol << 2);
#pragma unroll
      for (int nt = 0; nt < 4; ++nt) {
        const v16h b = *(const v16h*)(Bp + ((((size_t)(ntBase + nt)) * kTiles + ktile) << 9)
                                         + ((size_t)lane << 4));
#pragma unroll
        for (int mt = 0; mt < 4; ++mt) {
          acc[mt][nt] = __builtin_amdgcn_wmma_f32_16x16x32_f16(
              false, a[mt].h, false, b, (short)0, acc[mt][nt], false, false);
        }
      }
    }
    __syncthreads();
  }

  // Epilogue: C/D layout -> n = lane&15, m = 8*(lane>>4) + vgpr.
  const int nCol0 = nBase + (wCol << 6) + (lane & 15);
  const int mRow0 = mBase + wRow * 64 + ((lane >> 4) << 3);
#pragma unroll
  for (int mt = 0; mt < 4; ++mt) {
#pragma unroll
    for (int nt = 0; nt < 4; ++nt) {
      const int n = nCol0 + (nt << 4);
      const float bn = bias[n];
#pragma unroll
      for (int v = 0; v < 8; ++v) {
        const int m = mRow0 + (mt << 4) + v;
        const size_t o = (size_t)m * N + n;
        const float val = acc[mt][nt][v] + bn;
        if (outH) outH[o] = (_Float16)val;
        else      outF[o] = resid[o] + val;
      }
    }
  }
}

// ---------------------------------------------------------------------------
// Broadcast g_tok (NT,D) -> gbuf (B,NT,D).
// ---------------------------------------------------------------------------
__global__ void bcast_g(const float* __restrict__ g_tok, float* __restrict__ gbuf,
                        int n, int total) {
  int i = blockIdx.x * 256 + threadIdx.x;
  if (i >= total) return;
  gbuf[i] = g_tok[i % n];
}

// ---------------------------------------------------------------------------
// Small fp32 GEMM: out[R,N] = act( A[R,K] @ W[K,N] + bias ) (+ resid).
// One thread per output element (R is tiny: 32 rows).
// ---------------------------------------------------------------------------
__global__ void sgemm_small(const float* __restrict__ A, const float* __restrict__ W,
                            const float* __restrict__ bias, const float* __restrict__ resid,
                            float* __restrict__ out, int R, int K, int N, int act) {
  int idx = blockIdx.x * 256 + threadIdx.x;
  if (idx >= R * N) return;
  int row = idx / N, n = idx - row * N;
  const float* a = A + (size_t)row * K;
  float acc = bias[n];
  for (int k = 0; k < K; ++k) acc = fmaf(a[k], W[(size_t)k * N + n], acc);
  if (act) acc = acc / (1.0f + __expf(-acc));   // SiLU
  if (resid) acc += resid[idx];
  out[idx] = acc;
}

// ---------------------------------------------------------------------------
// Stage-1 scores: S[b,g,t] = scale * dot(q1[b,g], k1h[b,t]).  Block per (b,t).
// ---------------------------------------------------------------------------
__global__ void scores1(const _Float16* __restrict__ k1, const float* __restrict__ q1,
                        float* __restrict__ S) {
  const int bt = blockIdx.x;
  const int b = bt / Tc, t = bt - b * Tc;
  const int tid = threadIdx.x;
  const v4h kv = *(const v4h*)(k1 + (size_t)bt * Dc + tid * 4);
  const float kf0 = (float)kv[0], kf1 = (float)kv[1], kf2 = (float)kv[2], kf3 = (float)kv[3];
  __shared__ float red[NTc * 256];
#pragma unroll
  for (int g = 0; g < NTc; ++g) {
    const float* q = q1 + ((size_t)b * NTc + g) * Dc + tid * 4;
    red[g * 256 + tid] = kf0 * q[0] + kf1 * q[1] + kf2 * q[2] + kf3 * q[3];
  }
  __syncthreads();
  for (int s = 128; s > 0; s >>= 1) {
    if (tid < s) {
#pragma unroll
      for (int g = 0; g < NTc; ++g) red[g * 256 + tid] += red[g * 256 + tid + s];
    }
    __syncthreads();
  }
  if (tid < NTc) S[((size_t)b * NTc + tid) * Tc + t] = red[tid * 256] * SCALEc;
}

// ---------------------------------------------------------------------------
// Softmax over last axis (T) of (B*NT, T).
// ---------------------------------------------------------------------------
__global__ void softmax_rows(float* __restrict__ S, int T) {
  const int row = blockIdx.x, tid = threadIdx.x;
  float* s = S + (size_t)row * T;
  __shared__ float red[256];
  float m = -3.4e38f;
  for (int t = tid; t < T; t += 256) m = fmaxf(m, s[t]);
  red[tid] = m; __syncthreads();
  for (int st = 128; st > 0; st >>= 1) { if (tid < st) red[tid] = fmaxf(red[tid], red[tid + st]); __syncthreads(); }
  m = red[0]; __syncthreads();
  float sum = 0.f;
  for (int t = tid; t < T; t += 256) { const float e = __expf(s[t] - m); s[t] = e; sum += e; }
  red[tid] = sum; __syncthreads();
  for (int st = 128; st > 0; st >>= 1) { if (tid < st) red[tid] += red[tid + st]; __syncthreads(); }
  const float inv = 1.0f / red[0];
  for (int t = tid; t < T; t += 256) s[t] *= inv;
}

// ---------------------------------------------------------------------------
// ctx1 partials: part[c, bg, d] = sum_{t in chunk c} attn[bg,t] * v1[b,t,d].
// Deterministic (no float atomics); reduced by ctx1_reduce.
// ---------------------------------------------------------------------------
__global__ void ctx1_partial(const float* __restrict__ attn, const _Float16* __restrict__ v1,
                             float* __restrict__ part) {
  const int bg = blockIdx.x;          // B*NT
  const int b  = bg >> 3;
  const int t0 = blockIdx.y * TCHUNK;
  const int tid = threadIdx.x;
  const float* a = attn + (size_t)bg * Tc + t0;
  float a0 = 0.f, a1 = 0.f, a2 = 0.f, a3 = 0.f;
  for (int t = 0; t < TCHUNK; ++t) {
    const float wv = a[t];
    const v4h vv = *(const v4h*)(v1 + ((size_t)b * Tc + t0 + t) * Dc + tid * 4);
    a0 += wv * (float)vv[0]; a1 += wv * (float)vv[1];
    a2 += wv * (float)vv[2]; a3 += wv * (float)vv[3];
  }
  float* p = part + ((size_t)blockIdx.y * (Bc * NTc) + bg) * Dc + tid * 4;
  p[0] = a0; p[1] = a1; p[2] = a2; p[3] = a3;
}

__global__ void ctx1_reduce(const float* __restrict__ part, float* __restrict__ ctx) {
  const int i = blockIdx.x * 256 + threadIdx.x;   // over B*NT*D
  float s = 0.f;
#pragma unroll
  for (int c = 0; c < NCH; ++c) s += part[(size_t)c * (Bc * NTc * Dc) + i];
  ctx[i] = s;
}

// ---------------------------------------------------------------------------
// Stage-2: per (b,t): softmax over NT=8 scores, ctx2 = sum_g p[g]*v2[b,g,:].
// ---------------------------------------------------------------------------
__global__ void attn2_ctx(const _Float16* __restrict__ q2, const float* __restrict__ k2,
                          const float* __restrict__ v2, _Float16* __restrict__ ctx) {
  const int bt = blockIdx.x;
  const int b = bt / Tc;
  const int tid = threadIdx.x;
  const v4h qv = *(const v4h*)(q2 + (size_t)bt * Dc + tid * 4);
  const float qf0 = (float)qv[0], qf1 = (float)qv[1], qf2 = (float)qv[2], qf3 = (float)qv[3];
  __shared__ float red[NTc * 256];
  __shared__ float sc[NTc];
#pragma unroll
  for (int g = 0; g < NTc; ++g) {
    const float* kr = k2 + ((size_t)b * NTc + g) * Dc + tid * 4;
    red[g * 256 + tid] = qf0 * kr[0] + qf1 * kr[1] + qf2 * kr[2] + qf3 * kr[3];
  }
  __syncthreads();
  for (int s = 128; s > 0; s >>= 1) {
    if (tid < s) {
#pragma unroll
      for (int g = 0; g < NTc; ++g) red[g * 256 + tid] += red[g * 256 + tid + s];
    }
    __syncthreads();
  }
  if (tid < NTc) sc[tid] = red[tid * 256] * SCALEc;
  __syncthreads();
  float m = sc[0];
#pragma unroll
  for (int g = 1; g < NTc; ++g) m = fmaxf(m, sc[g]);
  float w[NTc], sum = 0.f;
#pragma unroll
  for (int g = 0; g < NTc; ++g) { w[g] = __expf(sc[g] - m); sum += w[g]; }
  const float inv = 1.0f / sum;
  float a0 = 0.f, a1 = 0.f, a2 = 0.f, a3 = 0.f;
#pragma unroll
  for (int g = 0; g < NTc; ++g) {
    const float* vr = v2 + ((size_t)b * NTc + g) * Dc + tid * 4;
    const float ww = w[g] * inv;
    a0 += ww * vr[0]; a1 += ww * vr[1]; a2 += ww * vr[2]; a3 += ww * vr[3];
  }
  v4h o;
  o[0] = (_Float16)a0; o[1] = (_Float16)a1; o[2] = (_Float16)a2; o[3] = (_Float16)a3;
  *(v4h*)(ctx + (size_t)bt * Dc + tid * 4) = o;
}

// ---------------------------------------------------------------------------
extern "C" void kernel_launch(void* const* d_in, const int* in_sizes, int n_in,
                              void* d_out, int out_size, void* d_ws, size_t ws_size,
                              hipStream_t stream) {
  (void)in_sizes; (void)n_in; (void)out_size; (void)ws_size;
  const float* x     = (const float*)d_in[0];
  const float* g_tok = (const float*)d_in[1];
  const float* w_ns1 = (const float*)d_in[2];
  const float* w_ng1 = (const float*)d_in[3];
  const float* w_ngm = (const float*)d_in[4];
  const float* w_ns2 = (const float*)d_in[5];
  const float* w_ng2 = (const float*)d_in[6];
  const float* q1_w = (const float*)d_in[7];  const float* q1_b = (const float*)d_in[8];
  const float* k1_w = (const float*)d_in[9];  const float* k1_b = (const float*)d_in[10];
  const float* v1_w = (const float*)d_in[11]; const float* v1_b = (const float*)d_in[12];
  const float* o1_w = (const float*)d_in[13]; const float* o1_b = (const float*)d_in[14];
  const float* q2_w = (const float*)d_in[15]; const float* q2_b = (const float*)d_in[16];
  const float* k2_w = (const float*)d_in[17]; const float* k2_b = (const float*)d_in[18];
  const float* v2_w = (const float*)d_in[19]; const float* v2_b = (const float*)d_in[20];
  const float* o2_w = (const float*)d_in[21]; const float* o2_b = (const float*)d_in[22];
  const float* m1_w = (const float*)d_in[23]; const float* m1_b = (const float*)d_in[24];
  const float* m2_w = (const float*)d_in[25]; const float* m2_b = (const float*)d_in[26];
  float* out = (float*)d_out;

  // ---- workspace carve-out (aliased where lifetimes allow) ----
  char* ws = (char*)d_ws;
  size_t off = 0;
  auto alloc = [&](size_t bytes) -> char* {
    off = (off + 255) & ~(size_t)255;
    char* p = ws + off; off += bytes; return p;
  };
  const size_t BTD = (size_t)BT * Dc;
  _Float16* s1h = (_Float16*)alloc(BTD * 2);
  _Float16* s2h = (_Float16*)alloc(BTD * 2);
  _Float16* k1h = (_Float16*)alloc(BTD * 2);
  _Float16* v1h = (_Float16*)alloc(BTD * 2);
  _Float16* q2h   = s1h;   // s1h dead after v1 GEMM
  _Float16* ctx2h = k1h;   // k1h dead after scores1
  _Float16* wpk1 = (_Float16*)alloc((size_t)Dc * Dc * 2);
  _Float16* wpv1 = (_Float16*)alloc((size_t)Dc * Dc * 2);
  _Float16* wpq2 = (_Float16*)alloc((size_t)Dc * Dc * 2);
  _Float16* wpo2 = (_Float16*)alloc((size_t)Dc * Dc * 2);
  float* S    = (float*)alloc((size_t)Bc * NTc * Tc * 4);
  float* gbuf = (float*)alloc((size_t)Bc * NTc * Dc * 4);
  float* g1b  = (float*)alloc((size_t)Bc * NTc * Dc * 4);
  float* q1b  = (float*)alloc((size_t)Bc * NTc * Dc * 4);
  float* ctx1 = (float*)alloc((size_t)Bc * NTc * Dc * 4);
  float* gmb  = (float*)alloc((size_t)Bc * NTc * Dc * 4);
  float* g2b  = (float*)alloc((size_t)Bc * NTc * Dc * 4);
  float* k2b  = (float*)alloc((size_t)Bc * NTc * Dc * 4);
  float* v2b  = (float*)alloc((size_t)Bc * NTc * Dc * 4);
  float* h1   = (float*)alloc((size_t)Bc * NTc * Hc * 4);
  float* cpart = (float*)alloc((size_t)NCH * Bc * NTc * Dc * 4);

  const dim3 blk(256);
  const int wBlocks = (Dc * Dc + 255) / 256;
  const dim3 ggrid(BT / 128, Dc / 256);     // 128 x 4 blocks, block tile 128x256
  const int gElems  = Bc * NTc * Dc;
  const int gBlocks = (gElems + 255) / 256;

  // 1. pack big weights into WMMA frag order (f16)
  pack_weight<<<wBlocks, blk, 0, stream>>>(k1_w, wpk1, Dc, Dc);
  pack_weight<<<wBlocks, blk, 0, stream>>>(v1_w, wpv1, Dc, Dc);
  pack_weight<<<wBlocks, blk, 0, stream>>>(q2_w, wpq2, Dc, Dc);
  pack_weight<<<wBlocks, blk, 0, stream>>>(o2_w, wpo2, Dc, Dc);
  // 2. fused RMS(x) -> s1h, s2h (f16)
  rms_cast2<<<BT, blk, 0, stream>>>(x, w_ns1, w_ns2, s1h, s2h);
  // 3-4. big WMMA GEMMs: k1, v1
  gemm_wmma<<<ggrid, blk, 0, stream>>>(s1h, wpk1, k1_b, nullptr, k1h, nullptr, BT, Dc, Dc);
  gemm_wmma<<<ggrid, blk, 0, stream>>>(s1h, wpv1, v1_b, nullptr, v1h, nullptr, BT, Dc, Dc);
  // 5-7. global-token path: g, rms, q1
  bcast_g<<<gBlocks, blk, 0, stream>>>(g_tok, gbuf, NTc * Dc, gElems);
  rms_rows<<<Bc * NTc, blk, 0, stream>>>(gbuf, w_ng1, g1b);
  sgemm_small<<<gBlocks, blk, 0, stream>>>(g1b, q1_w, q1_b, nullptr, q1b, Bc * NTc, Dc, Dc, 0);
  // 8-9. stage-1 attention scores + softmax
  scores1<<<BT, blk, 0, stream>>>(k1h, q1b, S);
  softmax_rows<<<Bc * NTc, blk, 0, stream>>>(S, Tc);
  // 10. q2 GEMM (output reuses s1h)
  gemm_wmma<<<ggrid, blk, 0, stream>>>(s2h, wpq2, q2_b, nullptr, q2h, nullptr, BT, Dc, Dc);
  // 11. ctx1 = attn1 @ v1 (deterministic two-phase)
  ctx1_partial<<<dim3(Bc * NTc, NCH), blk, 0, stream>>>(S, v1h, cpart);
  ctx1_reduce<<<gBlocks, blk, 0, stream>>>(cpart, ctx1);
  // 12. g += ctx1 @ o1_w + o1_b
  sgemm_small<<<gBlocks, blk, 0, stream>>>(ctx1, o1_w, o1_b, gbuf, gbuf, Bc * NTc, Dc, Dc, 0);
  // 13-15. SiLU MLP on global tokens
  rms_rows<<<Bc * NTc, blk, 0, stream>>>(gbuf, w_ngm, gmb);
  sgemm_small<<<(Bc * NTc * Hc + 255) / 256, blk, 0, stream>>>(gmb, m1_w, m1_b, nullptr, h1,
                                                               Bc * NTc, Dc, Hc, 1);
  sgemm_small<<<gBlocks, blk, 0, stream>>>(h1, m2_w, m2_b, gbuf, gbuf, Bc * NTc, Hc, Dc, 0);
  // 16-17. stage-2 k2, v2 from updated g
  rms_rows<<<Bc * NTc, blk, 0, stream>>>(gbuf, w_ng2, g2b);
  sgemm_small<<<gBlocks, blk, 0, stream>>>(g2b, k2_w, k2_b, nullptr, k2b, Bc * NTc, Dc, Dc, 0);
  sgemm_small<<<gBlocks, blk, 0, stream>>>(g2b, v2_w, v2_b, nullptr, v2b, Bc * NTc, Dc, Dc, 0);
  // 18. stage-2 attention (8-wide softmax fused) -> ctx2h (reuses k1h)
  attn2_ctx<<<BT, blk, 0, stream>>>(q2h, k2b, v2b, ctx2h);
  // 19. out = x + ctx2 @ o2_w + o2_b  (WMMA GEMM with fp32 residual epilogue)
  gemm_wmma<<<ggrid, blk, 0, stream>>>(ctx2h, wpo2, o2_b, x, nullptr, out, BT, Dc, Dc);
}
